// VQVAE_69269232549942
// MI455X (gfx1250) — matrix operand order repaired and checked by reading, not compiled, not run on hardware
//
#include <hip/hip_runtime.h>
#include <hip/hip_bf16.h>

typedef __attribute__((ext_vector_type(16))) _Float16 v16h;
typedef __attribute__((ext_vector_type(8)))  _Float16 v8h;
typedef __attribute__((ext_vector_type(8)))  float    v8f;

// ---------------------------------------------------------------------------
// Layer-specialized implicit-GEMM conv / transposed-conv on
// v_wmma_f32_16x16x32_f16. All shape parameters are compile-time: spatial
// dims are powers of two -> index math folds to shifts/ands; bounds checks
// compare against immediates; unsigned k-decomposition avoids signed-division
// fixups. M = Cout (NT x 16 tiles/wave), N = pixels (16/wave, 8 waves/block),
// K in 32-chunks staged through a 128-K LDS f16 weight panel (272B row
// stride -> 16B-aligned ds_load_b128 fragments). B gather: pass 1 issues all
// 16 unconditional clamped loads, pass 2 converts+masks (one wait/fragment).
// ---------------------------------------------------------------------------
template <int KH, int KW, int STRIDE, int PAD, int TRANS, int NT,
          int CIN, int HIN, int WIN, int COUT, int HOUT, int WOUT>
__global__ void __launch_bounds__(256)
wmma_conv(const float* __restrict__ X, const float* __restrict__ W,
          const float* __restrict__ Bias, const float* __restrict__ Res,
          float* __restrict__ Y, int N, int doRelu)
{
    constexpr int KP   = 128;        // K panel staged in LDS
    constexpr int RS   = KP + 8;     // row stride (halves): 272B = 17*16B
    constexpr int Ktot = CIN * KH * KW;
    constexpr int KtM1 = Ktot - 1;
    constexpr int HWin = HIN * WIN;
    constexpr int HWout = HOUT * WOUT;
    __shared__ alignas(16) _Float16 sw[NT * 16 * RS];

    const int tid  = threadIdx.x;
    const int lane = tid & 31;
    const int wave = tid >> 5;
    const int hi   = lane >> 4;
    const int mrow = lane & 15;

    const int pixTiles = (N * HWout) >> 4;
    int pixTile = blockIdx.x * 8 + wave;
    if (pixTile >= pixTiles) pixTile = pixTiles - 1;   // barriers stay uniform

    const unsigned pix = (unsigned)((pixTile << 4) + (lane & 15));
    const int wo = (int)(pix % WOUT);                  // WOUT: power of two
    const int ho = (int)((pix / WOUT) % HOUT);
    const int nb = (int)(pix / (unsigned)HWout);

    const int inBase = nb * (CIN * HWin);              // < 2^31 everywhere
    const int ihB = TRANS ? (ho - (KH - 1 - PAD)) : (ho * STRIDE - PAD);
    const int iwB = TRANS ? (wo - (KW - 1 - PAD)) : (wo * STRIDE - PAD);

    v8f accs[NT];
#pragma unroll
    for (int nt = 0; nt < NT; ++nt) accs[nt] = (v8f){};

    for (int pb = 0; pb < Ktot; pb += KP) {
        __syncthreads();
        // ---- stage weight panel (branchless: clamp + select) ----
        constexpr int ELEMS = NT * 16 * KP;
#pragma unroll 2
        for (int idx = tid; idx < ELEMS; idx += 256) {
            const int co  = idx >> 7;            // idx / KP
            const int kk  = idx & (KP - 1);
            const int k   = pb + kk;
            const bool ok = (k < Ktot) & (co < COUT);
            const unsigned kc = (unsigned)(k < KtM1 ? k : KtM1);
            const int coc = co < COUT - 1 ? co : COUT - 1;
            const int kw  = (int)(kc % KW);
            const unsigned t2 = kc / KW;
            const int kh  = (int)(t2 % KH);
            const int ci  = (int)(t2 / KH);
            int widx;
            if (TRANS)      // weight (Cin,Cout,KH,KW), spatially flipped
                widx = ((ci * COUT + coc) * KH + (KH - 1 - kh)) * KW + (KW - 1 - kw);
            else            // weight (Cout,Cin,KH,KW)
                widx = ((coc * CIN + ci) * KH + kh) * KW + kw;
            const float wv = W[widx];
            sw[co * RS + kk] = (_Float16)(ok ? wv : 0.0f);
        }
        __syncthreads();

        const int kend = (pb + KP < Ktot) ? (pb + KP) : Ktot;
#pragma unroll 2
        for (int kc0 = pb; kc0 < kend; kc0 += 32) {
            // ---- B fragment pass 1: batched unconditional loads ----
            float xs[16];
            bool  vd[16];
#pragma unroll
            for (int i = 0; i < 16; ++i) {
                const int k = kc0 + hi * 16 + i;
                const unsigned kc = (unsigned)(k < KtM1 ? k : KtM1); // addr clamp
                const int kw = (int)(kc % KW);
                const unsigned t2 = kc / KW;
                const int kh = (int)(t2 % KH);
                const int ci = (int)(t2 / KH);
                int ih, iw;
                bool valid;
                if (!TRANS) {
                    ih = ihB + kh;
                    iw = iwB + kw;
                    valid = (ih >= 0) & (ih < HIN) & (iw >= 0) & (iw < WIN);
                } else {
                    const int ihd = ihB + kh;          // x2-dilated coords
                    const int iwd = iwB + kw;
                    ih = ihd >> 1;
                    iw = iwd >> 1;
                    valid = (ihd >= 0) & (iwd >= 0) & !(ihd & 1) & !(iwd & 1)
                          & (ih < HIN) & (iw < WIN);
                }
                const int ihc = ih < 0 ? 0 : (ih >= HIN ? HIN - 1 : ih);
                const int iwc = iw < 0 ? 0 : (iw >= WIN ? WIN - 1 : iw);
                vd[i] = valid;
                xs[i] = X[inBase + ci * HWin + ihc * WIN + iwc];
            }
            // ---- pass 2: convert + mask ----
            v16h bfrag;
#pragma unroll
            for (int i = 0; i < 16; ++i)
                bfrag[i] = (_Float16)(vd[i] ? xs[i] : 0.0f);

            // ---- A fragments from LDS; one wmma per cout tile ----
            const int kkc = kc0 - pb;
#pragma unroll
            for (int nt = 0; nt < NT; ++nt) {
                const int off = ((nt << 4) + mrow) * RS + kkc + hi * 16;
                const v8h alo = *(const v8h*)(&sw[off]);        // K = kc0+hi*16..
                const v8h ahi = *(const v8h*)(&sw[off + 8]);    // K = kc0+hi*16+8..
                const v16h afrag = __builtin_shufflevector(
                    alo, ahi, 0, 1, 2, 3, 4, 5, 6, 7, 8, 9, 10, 11, 12, 13, 14, 15);
                accs[nt] = __builtin_amdgcn_wmma_f32_16x16x32_f16(
                    false, afrag, false, bfrag, (short)0, accs[nt], false, false);
            }
        }
    }

    // ---- epilogue: C/D layout -> lane n = lane&15, M = r + 8*hi ----
    const int outBase = nb * (COUT * HWout) + ho * WOUT + wo;
#pragma unroll
    for (int nt = 0; nt < NT; ++nt) {
#pragma unroll
        for (int r = 0; r < 8; ++r) {
            const int co = (nt << 4) + r + 8 * hi;
            if (co < COUT) {              // constant-folded per instantiation
                float v = accs[nt][r];
                if (Bias) v += Bias[co];
                if (doRelu) v = fmaxf(v, 0.0f);
                const int oidx = outBase + co * HWout;
                if (Res) v += Res[oidx];
                Y[oidx] = v;
            }
        }
    }
}

// ---------------------------------------------------------------------------
// Vector quantization: per-pixel argmin over 512 codes (D=64), codebook tiled
// through LDS (4 x 32KB); writes q (NCHW) and accumulates sum((q-z)^2).
// ---------------------------------------------------------------------------
__global__ void __launch_bounds__(256)
vq_kernel(const float* __restrict__ z, const float* __restrict__ cb,
          float* __restrict__ q, float* __restrict__ commitAcc)
{
    constexpr int HW = 4096;
    __shared__ float scb[128 * 64];
    __shared__ float sred[256];
    const int tid = threadIdx.x;
    const unsigned p = blockIdx.x * 256 + tid;
    const int n   = (int)(p / HW);
    const int hw  = (int)(p % HW);
    const int zb  = n * 64 * HW + hw;

    float zv[64];
#pragma unroll
    for (int d = 0; d < 64; ++d)
        zv[d] = z[zb + d * HW];

    float best = 3.4e38f;
    int bestIdx = 0;
    for (int tile = 0; tile < 4; ++tile) {
        __syncthreads();
        for (int i = tid; i < 128 * 64; i += 256)
            scb[i] = cb[tile * (128 * 64) + i];
        __syncthreads();
        for (int c = 0; c < 128; ++c) {
            float d2 = 0.0f;
#pragma unroll
            for (int d = 0; d < 64; ++d) {
                float df = zv[d] - scb[c * 64 + d];
                d2 = fmaf(df, df, d2);
            }
            if (d2 < best) { best = d2; bestIdx = tile * 128 + c; }
        }
    }

    float csum = 0.0f;
#pragma unroll
    for (int d = 0; d < 64; ++d) {
        float qd = cb[bestIdx * 64 + d];
        float df = qd - zv[d];
        csum = fmaf(df, df, csum);
        q[zb + d * HW] = qd;
    }

    sred[tid] = csum;
    __syncthreads();
    for (int s = 128; s > 0; s >>= 1) {
        if (tid < s) sred[tid] += sred[tid + s];
        __syncthreads();
    }
    if (tid == 0) atomicAdd(commitAcc, sred[0]);
}

// ---------------------------------------------------------------------------
__global__ void __launch_bounds__(256)
sqdiff_kernel(const float* __restrict__ a, const float* __restrict__ b,
              int n, float* __restrict__ acc)
{
    __shared__ float sred[256];
    int i0 = blockIdx.x * 256 + threadIdx.x;
    int stride = gridDim.x * 256;
    float s = 0.0f;
    for (int i = i0; i < n; i += stride) {
        float d = a[i] - b[i];
        s = fmaf(d, d, s);
    }
    sred[threadIdx.x] = s;
    __syncthreads();
    for (int st = 128; st > 0; st >>= 1) {
        if (threadIdx.x < st) sred[threadIdx.x] += sred[threadIdx.x + st];
        __syncthreads();
    }
    if (threadIdx.x == 0) atomicAdd(acc, sred[0]);
}

__global__ void zero_acc_kernel(float* acc) { acc[0] = 0.0f; acc[1] = 0.0f; }

__global__ void finalize_kernel(const float* __restrict__ acc, float* __restrict__ out2)
{
    float recMean    = acc[1] / 3145728.0f;   // 16*3*256*256
    float commitMean = acc[0] / 4194304.0f;   // 65536 px * 64 dims
    out2[0] = recMean + 1.25f * commitMean;   // total_loss = rec + (1+0.25)*mse(q,z)
    out2[1] = recMean;                        // reconstruction_loss
}

// ---------------------------------------------------------------------------
template <int KH, int KW, int STRIDE, int PAD, int TRANS, int NT,
          int CIN, int HIN, int WIN, int COUT, int HOUT, int WOUT>
static void launch_conv(hipStream_t stream,
                        const float* X, const float* W, const float* B, const float* R,
                        float* Y, int N, int relu)
{
    int pixTiles = (N * HOUT * WOUT) / 16;
    dim3 grid((pixTiles + 7) / 8, 1);
    wmma_conv<KH, KW, STRIDE, PAD, TRANS, NT, CIN, HIN, WIN, COUT, HOUT, WOUT>
        <<<grid, 256, 0, stream>>>(X, W, B, R, Y, N, relu);
}

extern "C" void kernel_launch(void* const* d_in, const int* in_sizes, int n_in,
                              void* d_out, int out_size, void* d_ws, size_t ws_size,
                              hipStream_t stream)
{
    (void)in_sizes; (void)n_in; (void)out_size; (void)ws_size;

    const float* inputs     = (const float*)d_in[0];
    const float* enc_w1     = (const float*)d_in[1];
    const float* enc_b1     = (const float*)d_in[2];
    const float* enc_w2     = (const float*)d_in[3];
    const float* enc_b2     = (const float*)d_in[4];
    const float* enc_w3     = (const float*)d_in[5];
    const float* enc_b3     = (const float*)d_in[6];
    const float* enc_res_w1 = (const float*)d_in[7];    // (3,64,128,3,3)
    const float* enc_res_w2 = (const float*)d_in[8];    // (3,128,64,1,1)
    const float* vq_w       = (const float*)d_in[9];
    const float* vq_b       = (const float*)d_in[10];
    const float* codebook   = (const float*)d_in[11];   // (512,64)
    const float* dec_w1     = (const float*)d_in[12];
    const float* dec_b1     = (const float*)d_in[13];
    const float* dec_res_w1 = (const float*)d_in[14];   // (3,32,64,3,3)
    const float* dec_res_w2 = (const float*)d_in[15];   // (3,64,32,1,1)
    const float* dec_w2t    = (const float*)d_in[16];   // (64,32,4,4)
    const float* dec_b2     = (const float*)d_in[17];
    const float* dec_w3t    = (const float*)d_in[18];   // (32,3,4,4)
    const float* dec_b3     = (const float*)d_in[19];

    float* ws   = (float*)d_ws;
    float* acc  = ws;                       // acc[0]=commit sum, acc[1]=rec sum
    float* buf0 = ws + 256;                 // 16.78M floats
    float* buf1 = buf0 + 16777216;          //  8.39M floats
    float* buf2 = buf1 + 8388608;           //  8.39M floats
    float* buf3 = buf2 + 8388608;           //  4.19M floats
    float* rec  = (float*)d_out;

    zero_acc_kernel<<<1, 1, 0, stream>>>(acc);

    // ---------------- Encoder ----------------
    launch_conv<4,4,2,1,0,4,   3,256,256,  64,128,128>(stream, inputs, enc_w1, enc_b1, nullptr, buf0, 16, 0);
    launch_conv<4,4,2,1,0,8,  64,128,128, 128, 64, 64>(stream, buf0,   enc_w2, enc_b2, nullptr, buf1, 16, 0);
    launch_conv<3,3,1,1,0,8, 128, 64, 64, 128, 64, 64>(stream, buf1,   enc_w3, enc_b3, nullptr, buf2, 16, 0);

    // Encoder resblocks: x + relu(conv1x1(relu(conv3x3(x))))
    const long eW1 = 64L * 128 * 9, eW2 = 128L * 64;
    launch_conv<3,3,1,1,0,4, 128,64,64,  64,64,64>(stream, buf2, enc_res_w1 + 0 * eW1, nullptr, nullptr, buf3, 16, 1);
    launch_conv<1,1,1,0,0,8,  64,64,64, 128,64,64>(stream, buf3, enc_res_w2 + 0 * eW2, nullptr, buf2,    buf1, 16, 1);
    launch_conv<3,3,1,1,0,4, 128,64,64,  64,64,64>(stream, buf1, enc_res_w1 + 1 * eW1, nullptr, nullptr, buf3, 16, 1);
    launch_conv<1,1,1,0,0,8,  64,64,64, 128,64,64>(stream, buf3, enc_res_w2 + 1 * eW2, nullptr, buf1,    buf2, 16, 1);
    launch_conv<3,3,1,1,0,4, 128,64,64,  64,64,64>(stream, buf2, enc_res_w1 + 2 * eW1, nullptr, nullptr, buf3, 16, 1);
    launch_conv<1,1,1,0,0,8,  64,64,64, 128,64,64>(stream, buf3, enc_res_w2 + 2 * eW2, nullptr, buf2,    buf1, 16, 1);

    // Pre-VQ projection -> z (buf3)
    launch_conv<1,1,1,0,0,4, 128,64,64, 64,64,64>(stream, buf1, vq_w, vq_b, nullptr, buf3, 16, 0);

    // ---------------- Vector quantization ----------------
    vq_kernel<<<256, 256, 0, stream>>>(buf3, codebook, buf0, acc + 0);

    // ---------------- Decoder ----------------
    launch_conv<3,3,1,1,0,4, 64,64,64, 64,64,64>(stream, buf0, dec_w1, dec_b1, nullptr, buf2, 16, 0);

    const long dW1 = 32L * 64 * 9, dW2 = 64L * 32;
    launch_conv<3,3,1,1,0,2, 64,64,64, 32,64,64>(stream, buf2, dec_res_w1 + 0 * dW1, nullptr, nullptr, buf3, 16, 1);
    launch_conv<1,1,1,0,0,4, 32,64,64, 64,64,64>(stream, buf3, dec_res_w2 + 0 * dW2, nullptr, buf2,    buf1, 16, 1);
    launch_conv<3,3,1,1,0,2, 64,64,64, 32,64,64>(stream, buf1, dec_res_w1 + 1 * dW1, nullptr, nullptr, buf3, 16, 1);
    launch_conv<1,1,1,0,0,4, 32,64,64, 64,64,64>(stream, buf3, dec_res_w2 + 1 * dW2, nullptr, buf1,    buf2, 16, 1);
    launch_conv<3,3,1,1,0,2, 64,64,64, 32,64,64>(stream, buf2, dec_res_w1 + 2 * dW1, nullptr, nullptr, buf3, 16, 1);
    launch_conv<1,1,1,0,0,4, 32,64,64, 64,64,64>(stream, buf3, dec_res_w2 + 2 * dW2, nullptr, buf2,    buf1, 16, 1);

    // Transposed convs (x2 lhs-dilated gather, flipped (Cin,Cout,k,k) weights)
    launch_conv<4,4,2,1,1,2, 64, 64, 64, 32,128,128>(stream, buf1, dec_w2t, dec_b2, nullptr, buf0, 16, 0);
    launch_conv<4,4,2,1,1,1, 32,128,128,  3,256,256>(stream, buf0, dec_w3t, dec_b3, nullptr, rec,  16, 0);

    // ---------------- Losses ----------------
    sqdiff_kernel<<<1024, 256, 0, stream>>>(rec, inputs, 16 * 3 * 256 * 256, acc + 1);
    finalize_kernel<<<1, 1, 0, stream>>>(acc, rec + 16L * 3 * 256 * 256);
}
